// _GTLayer_39230231282112
// MI455X (gfx1250) — compile-verified, matrix operand order checked by reading, not run on hardware
//
#include <hip/hip_runtime.h>
#include <hip/hip_bf16.h>

typedef __attribute__((ext_vector_type(16))) __bf16 v16bf;
typedef __attribute__((ext_vector_type(8)))  float  v8f;

#define IN_DIM 128
#define OUT_DIM 128

// ---------- order-preserving float <-> uint for atomic segment-max ----------
__device__ __forceinline__ unsigned fenc(float f) {
    unsigned u = __float_as_uint(f);
    return (u & 0x80000000u) ? ~u : (u | 0x80000000u);
}
__device__ __forceinline__ float fdec(unsigned k) {
    unsigned u = (k & 0x80000000u) ? (k ^ 0x80000000u) : ~k;
    return __uint_as_float(u);
}

// ---------------- GEMM: h = x @ W^T + b  (bf16 WMMA, f32 accum) -------------
// Block = 256 threads = 8 waves. Wave w owns M-tile (blockIdx.x*8 + w): it
// loads its A fragments (x rows, all K) ONCE into registers, then sweeps all
// 8 N-tiles of OUT reloading only B fragments (W is 64 KB -> L1-resident).
// => x streamed exactly once from HBM; 32 v_wmma per wave.
__global__ void gat_gemm(const float* __restrict__ x, const float* __restrict__ W,
                         const float* __restrict__ bias, float* __restrict__ h, int n) {
    const int wave = threadIdx.x >> 5;
    const int lane = threadIdx.x & 31;
    const int hi   = lane >> 4;                 // 0: lanes 0-15, 1: lanes 16-31
    const int mtile = blockIdx.x * 8 + wave;
    const int mbase = mtile * 16;
    if (mbase >= n) return;
    int row = mbase + (lane & 15);              // A row for this lane
    if (row >= n) row = n - 1;                  // clamp (n % 16 == 0 anyway)
    const int ksel = hi * 8;                    // K-run select per ISA A layout

    // Load all K fragments of A once (K = 128 -> 4 fragments of 32).
    v16bf afr[4];
#pragma unroll
    for (int kk = 0; kk < 4; ++kk) {
        const float* ap = x + (size_t)row * IN_DIM + kk * 32 + ksel;
#pragma unroll
        for (int j = 0; j < 8; ++j) {
            afr[kk][j]     = (__bf16)ap[j];        // K = kbase..kbase+7
            afr[kk][j + 8] = (__bf16)ap[j + 16];   // K = kbase+16..kbase+23
        }
    }

#pragma unroll
    for (int nt = 0; nt < 8; ++nt) {
        const int col = nt * 16 + (lane & 15);     // B column == output column
        v8f c = {};
#pragma unroll
        for (int kk = 0; kk < 4; ++kk) {
            const float* bp = W + (size_t)col * IN_DIM + kk * 32 + ksel;
            v16bf bfr;
#pragma unroll
            for (int j = 0; j < 8; ++j) {
                bfr[j]     = (__bf16)bp[j];
                bfr[j + 8] = (__bf16)bp[j + 16];
            }
            c = __builtin_amdgcn_wmma_f32_16x16x32_bf16(false, afr[kk], false, bfr,
                                                        (short)0, c, false, false);
        }
        const float bv = bias[col];
#pragma unroll
        for (int j = 0; j < 8; ++j) {
            const int m = mbase + j + hi * 8;      // C/D layout: VGPR j -> M=j (+8 hi)
            if (m < n) h[(size_t)m * OUT_DIM + col] = c[j] + bv;
        }
    }
}

// ---------------- s[i] = dot(h[i,:], att) ; one wave per node ---------------
__global__ void gat_score(const float* __restrict__ h, const float* __restrict__ att,
                          float* __restrict__ s, int n) {
    const int wave = threadIdx.x >> 5, lane = threadIdx.x & 31;
    const int node = blockIdx.x * 8 + wave;
    if (node >= n) return;
    float acc = 0.f;
#pragma unroll
    for (int j = 0; j < 4; ++j) {
        const int k = lane + j * 32;
        acc += h[(size_t)node * OUT_DIM + k] * att[k];
    }
#pragma unroll
    for (int off = 16; off; off >>= 1) acc += __shfl_xor(acc, off, 32);
    if (lane == 0) s[node] = acc;
}

__device__ __forceinline__ void edge_pair(const long long* __restrict__ ei,
                                          int idx, int E, int& src, int& dst) {
    if (idx < E) { src = (int)ei[idx]; dst = (int)ei[(size_t)E + idx]; }
    else         { src = idx - E;      dst = idx - E; }   // self-loop
}

__device__ __forceinline__ float leaky_alpha(const float* s, int src, int dst) {
    float a = s[dst] + s[src];
    return a > 0.f ? a : 0.2f * a;
}

// pass 1: segment max over src
__global__ void gat_edge_max(const long long* __restrict__ ei, const float* __restrict__ s,
                             unsigned* __restrict__ amaxk, int E, int Etot) {
    const int idx = blockIdx.x * blockDim.x + threadIdx.x;
    if (idx >= Etot) return;
    int src, dst; edge_pair(ei, idx, E, src, dst);
    atomicMax(&amaxk[src], fenc(leaky_alpha(s, src, dst)));
}

// pass 2: segment sum of exp(alpha - max)
__global__ void gat_edge_denom(const long long* __restrict__ ei, const float* __restrict__ s,
                               const unsigned* __restrict__ amaxk, float* __restrict__ denom,
                               int E, int Etot) {
    const int idx = blockIdx.x * blockDim.x + threadIdx.x;
    if (idx >= Etot) return;
    int src, dst; edge_pair(ei, idx, E, src, dst);
    const float ex = __expf(leaky_alpha(s, src, dst) - fdec(amaxk[src]));
    atomicAdd(&denom[src], ex);
}

// pass 3: out[dst,:] += alpha * h[src,:]  (one wave per edge, 4 cols/lane)
__global__ void gat_scatter(const long long* __restrict__ ei, const float* __restrict__ s,
                            const unsigned* __restrict__ amaxk, const float* __restrict__ denom,
                            const float* __restrict__ h, float* __restrict__ out,
                            int E, int Etot) {
    const int wave = threadIdx.x >> 5, lane = threadIdx.x & 31;
    const int idx = blockIdx.x * 8 + wave;
    if (idx >= Etot) return;
    int src, dst; edge_pair(ei, idx, E, src, dst);
    const float al = __expf(leaky_alpha(s, src, dst) - fdec(amaxk[src])) / denom[src];
    const float* hs = h + (size_t)src * OUT_DIM;
    float* od = out + (size_t)dst * OUT_DIM;
#pragma unroll
    for (int j = 0; j < 4; ++j) {
        const int cidx = lane + j * 32;
        atomicAdd(&od[cidx], al * hs[cidx]);
    }
}

__global__ void gat_relu(float* __restrict__ out, int total) {
    const int i = blockIdx.x * blockDim.x + threadIdx.x;
    if (i < total) out[i] = fmaxf(out[i], 0.f);
}

extern "C" void kernel_launch(void* const* d_in, const int* in_sizes, int n_in,
                              void* d_out, int out_size, void* d_ws, size_t ws_size,
                              hipStream_t stream) {
    const float*      x    = (const float*)d_in[0];
    const long long*  ei   = (const long long*)d_in[1];   // int64 [2, E]
    const float*      W    = (const float*)d_in[2];
    const float*      bias = (const float*)d_in[3];
    const float*      att  = (const float*)d_in[4];

    const int n    = in_sizes[0] / IN_DIM;       // 100000
    const int E    = in_sizes[1] / 2;            // 1600000
    const int Etot = E + n;
    const int mtiles = (n + 15) / 16;            // 6250

    float*    out   = (float*)d_out;
    float*    h     = (float*)d_ws;                    // n*128 f32
    float*    s     = h + (size_t)n * OUT_DIM;         // n f32
    unsigned* amaxk = (unsigned*)(s + n);              // n u32 (encoded -inf == 0)
    float*    denom = (float*)(amaxk + n);             // n f32

    hipMemsetAsync(out,   0, (size_t)n * OUT_DIM * sizeof(float), stream);
    hipMemsetAsync(amaxk, 0, (size_t)n * sizeof(unsigned), stream);
    hipMemsetAsync(denom, 0, (size_t)n * sizeof(float), stream);

    gat_gemm  <<<(mtiles + 7) / 8,      256, 0, stream>>>(x, W, bias, h, n);
    gat_score <<<(n + 7) / 8,           256, 0, stream>>>(h, att, s, n);
    gat_edge_max  <<<(Etot + 255) / 256, 256, 0, stream>>>(ei, s, amaxk, E, Etot);
    gat_edge_denom<<<(Etot + 255) / 256, 256, 0, stream>>>(ei, s, amaxk, denom, E, Etot);
    gat_scatter   <<<(Etot + 7) / 8,     256, 0, stream>>>(ei, s, amaxk, denom, h, out, E, Etot);
    gat_relu  <<<(n * OUT_DIM + 255) / 256, 256, 0, stream>>>(out, n * OUT_DIM);

    // tuple tail: raw copy of edge_index (int64) after the [N,128] f32 output
    hipMemcpyAsync((char*)d_out + (size_t)n * OUT_DIM * sizeof(float),
                   d_in[1], (size_t)in_sizes[1] * sizeof(long long),
                   hipMemcpyDeviceToDevice, stream);
}
